// LongTermMemoryModule_41145786695766
// MI455X (gfx1250) — matrix operand order repaired and checked by reading, not compiled, not run on hardware
//
#include <hip/hip_runtime.h>
#include <hip/hip_bf16.h>
#include <math.h>

static constexpr int S   = 1024;
static constexpr int B   = 32;
static constexpr int D   = 512;          // D == KD == VD == QD == HID
static constexpr int SB  = S * B;
static constexpr int NBLK  = 64;         // persistent scan blocks
static constexpr int WPB   = 8;          // waves per 256-thread block
static constexpr int NWAVE = NBLK * WPB; // 512 waves
static constexpr float DPRED_SCALE = 2.0f / (float)(B * D);

typedef __attribute__((ext_vector_type(16))) __bf16 v16bf;
typedef __attribute__((ext_vector_type(8)))  float  v8f;
typedef int v4i_vs __attribute__((vector_size(16)));

union Frag16 { v16bf v; uint4 q[2]; };

__device__ __forceinline__ __bf16 to_bf(float f) {
  union { float f; unsigned u; } v; v.f = f;
  unsigned r = v.u + 0x7FFFu + ((v.u >> 16) & 1u);
  union { unsigned short s; __bf16 b; } o; o.s = (unsigned short)(r >> 16);
  return o.b;
}

__device__ __forceinline__ float sigmoidf_(float x) { return 1.0f / (1.0f + expf(-x)); }
__device__ __forceinline__ float softplusf_(float x) { return (x > 20.0f) ? x : log1pf(expf(x)); }

__device__ __forceinline__ float wave_reduce_add(float v) {
#pragma unroll
  for (int m = 16; m > 0; m >>= 1) v += __shfl_xor(v, m, 32);
  return v;
}

// ---- CDNA5 async global->LDS copy (ASYNCcnt-tracked) with builtin/asm fallback.
__device__ __forceinline__ void async_ld_b128(const void* gsrc, void* lds_dst) {
#if __has_builtin(__builtin_amdgcn_global_load_async_to_lds_b128)
  __builtin_amdgcn_global_load_async_to_lds_b128(
      (__attribute__((address_space(1))) v4i_vs*)(void*)gsrc,
      (__attribute__((address_space(3))) v4i_vs*)lds_dst, 0, 0);
#else
  const unsigned loff =
      (unsigned)(size_t)(__attribute__((address_space(3))) char*)lds_dst;
  asm volatile("global_load_async_to_lds_b128 %0, %1, off"
               :: "v"(loff), "v"(gsrc)
               : "memory");
#endif
}

__device__ __forceinline__ void wait_async0() {
#if __has_builtin(__builtin_amdgcn_s_wait_asynccnt)
  __builtin_amdgcn_s_wait_asynccnt(0);
#else
  asm volatile("s_wait_asynccnt 0x0" ::: "memory");
#endif
}

// A fragment (16xK tile, K-chunk of 32) from row-major bf16 [rows x ld].
// Lane l<16 holds row l, K in {0..7, 16..23}; lane>=16 holds row l-16, K in {8..15, 24..31}.
__device__ __forceinline__ v16bf load_frag_A(const __bf16* base, int ld, int row0, int k0, int lane) {
  Frag16 f;
  const __bf16* rp = base + (size_t)(row0 + (lane & 15)) * ld + k0 + ((lane >> 4) << 3);
  f.q[0] = *(const uint4*)(rp);
  f.q[1] = *(const uint4*)(rp + 16);
  return f.v;
}

// B fragment (32x16, B[k][n] = M[n0+n][k0+k]) from row-major bf16 M — the "@ M^T" form.
// Lane group 0 holds K in {0..7, 16..23}, group 1 holds K in {8..15, 24..31}.
__device__ __forceinline__ v16bf load_frag_Bt(const __bf16* base, int ld, int n0, int k0, int lane) {
  Frag16 f;
  const __bf16* rp = base + (size_t)(n0 + (lane & 15)) * ld + k0 + ((lane >> 4) << 3);
  f.q[0] = *(const uint4*)(rp);
  f.q[1] = *(const uint4*)(rp + 16);
  return f.v;
}

__device__ __forceinline__ v8f wmma_bf16(v16bf a, v16bf b, v8f c) {
  return __builtin_amdgcn_wmma_f32_16x16x32_bf16(false, a, false, b, (short)0, c, false, false);
}

__device__ __forceinline__ void grid_sync(unsigned* bar, unsigned& gen) {
  __threadfence();
  __syncthreads();
  if (threadIdx.x == 0) {
    gen += 1;
    atomicAdd(bar, 1u);
    volatile unsigned* vb = bar;
    const unsigned target = gen * (unsigned)NBLK;
    while (*vb < target) __builtin_amdgcn_s_sleep(1);
  }
  __syncthreads();
  __threadfence();
}

// ---------------------------------------------------------------- utility kernels

__global__ void __launch_bounds__(256) f2bf_kernel(const float* __restrict__ in,
                                                   __bf16* __restrict__ out, size_t n) {
  for (size_t i = blockIdx.x * (size_t)blockDim.x + threadIdx.x; i < n;
       i += (size_t)gridDim.x * blockDim.x)
    out[i] = to_bf(in[i]);
}

// C[m,n] = sum_k A[m,k] * W[n,k] + bias[n].
// Block = 8 waves: one shared n-tile, 8 consecutive m-tiles. The 16-row weight
// panel (16 x K bf16 = 16KB) is staged in LDS with async copies and reused by
// all 8 waves (8x reduction in B-operand traffic); B fragments come from LDS.
__global__ void __launch_bounds__(256) gemm_bias_wt(const __bf16* __restrict__ A,
                                                    const __bf16* __restrict__ W,
                                                    const float* __restrict__ bias,
                                                    float* __restrict__ C,
                                                    int M, int N, int K) {
  __shared__ __bf16 wtile[16 * D];  // K == D for all uses here
  const int lane = threadIdx.x & 31;
  const int wv   = threadIdx.x >> 5;
  const int ntiles = N >> 4;
  const int nt = blockIdx.x % ntiles;
  const int mb = blockIdx.x / ntiles;        // 128-row block
  const int mt = mb * 8 + wv;

  // async-stage W rows [nt*16, nt*16+16) x K into LDS: 1024 x 16B, 4 per thread
  {
    const __bf16* src = W + (size_t)nt * 16 * K;
    const int chunks = (16 * K) / 8;  // 8 bf16 = 16 bytes per chunk
    for (int i = threadIdx.x; i < chunks; i += 256)
      async_ld_b128(src + (size_t)i * 8, &wtile[i * 8]);
  }
  wait_async0();
  __syncthreads();

  v8f acc = {0.f, 0.f, 0.f, 0.f, 0.f, 0.f, 0.f, 0.f};
  for (int k0 = 0; k0 < K; k0 += 32) {
    __builtin_prefetch(A + (size_t)(mt * 16 + (lane & 15)) * K + k0 + 32, 0, 1);
    v16bf a = load_frag_A(A, K, mt * 16, k0, lane);
    v16bf b = load_frag_Bt(wtile, K, 0, k0, lane);  // LDS-resident panel
    acc = wmma_bf16(a, b, acc);
  }
  const int col = nt * 16 + (lane & 15);
  const int rbase = mt * 16 + ((lane >> 4) << 3);
  const float bv = bias[col];
#pragma unroll
  for (int r = 0; r < 8; ++r)
    C[(size_t)(rbase + r) * N + col] = acc[r] + bv;
}

// alpha/eta/theta per step: mean over batch of sigmoid/sigmoid/softplus of x@w + b.
__global__ void __launch_bounds__(256) gate_kernel(const float* __restrict__ x,
                                                   const float* __restrict__ aw, const float* __restrict__ ab,
                                                   const float* __restrict__ ew, const float* __restrict__ eb,
                                                   const float* __restrict__ tw, const float* __restrict__ tb,
                                                   float* __restrict__ gates) {
  const int gw = blockIdx.x * (blockDim.x >> 5) + (threadIdx.x >> 5);
  const int lane = threadIdx.x & 31;
  if (gw >= SB) return;
  const float* xr = x + (size_t)gw * D;
  float da = 0.f, de = 0.f, dt = 0.f;
#pragma unroll
  for (int ii = 0; ii < 16; ++ii) {
    const int c = ii * 32 + lane;
    const float xv = xr[c];
    da += xv * aw[c]; de += xv * ew[c]; dt += xv * tw[c];
  }
  da = wave_reduce_add(da); de = wave_reduce_add(de); dt = wave_reduce_add(dt);
  if (lane == 0) {
    const int s = gw >> 5;
    atomicAdd(&gates[s],         sigmoidf_(da + ab[0]) * (1.0f / B));
    atomicAdd(&gates[S + s],     sigmoidf_(de + eb[0]) * (1.0f / B));
    atomicAdd(&gates[2 * S + s], softplusf_(dt + tb[0]) * (1.0f / B));
  }
}

__global__ void __launch_bounds__(256) init_state(const float* __restrict__ MW1, const float* __restrict__ Mb1,
                                                  const float* __restrict__ MW2, const float* __restrict__ Mb2,
                                                  float* W1, float* b1, float* W2, float* b2,
                                                  float* S1, float* Sb1, float* S2, float* Sb2,
                                                  __bf16* W1bf, __bf16* W2bf, __bf16* W2bfT) {
  for (size_t i = blockIdx.x * (size_t)blockDim.x + threadIdx.x; i < (size_t)D * D;
       i += (size_t)gridDim.x * blockDim.x) {
    const int row = (int)(i >> 9), col = (int)(i & (D - 1));
    const float w1 = MW1[i], w2 = MW2[i];
    W1[i] = w1; S1[i] = 0.f; W1bf[i] = to_bf(w1);
    W2[i] = w2; S2[i] = 0.f; W2bf[i] = to_bf(w2);
    W2bfT[(size_t)col * D + row] = to_bf(w2);
    if (i < (size_t)D) { b1[i] = Mb1[i]; b2[i] = Mb2[i]; Sb1[i] = 0.f; Sb2[i] = 0.f; }
  }
}

// ---------------------------------------------------------------- persistent scan

struct ScanArgs {
  const float *keys_pre, *values_pre, *queries_pre;
  const float *ck_w, *ck_b, *cv_w, *cv_b, *cq_w, *cq_b;
  const float *gates;                       // [3*S]: alpha, eta, theta
  float *W1, *W2, *S1, *S2, *b1, *b2, *Sb1, *Sb2;
  __bf16 *W1bf, *W2bf, *W2bfT;
  __bf16 *kt_bf, *ktT_bf, *qt_bf, *h_bf, *hT_bf;
  __bf16 *dpred_bf, *dpredT_bf, *dpreT_bf, *hq_bf;
  float *vt, *sp, *db1, *db2;
  unsigned *bar;
  float *out;
};

__global__ void __launch_bounds__(256) scan_kernel(ScanArgs P) {
  const int lane = threadIdx.x & 31;
  const int gw = blockIdx.x * WPB + (threadIdx.x >> 5);
  unsigned gen = 0;
  const v8f Z8 = {0.f, 0.f, 0.f, 0.f, 0.f, 0.f, 0.f, 0.f};

  for (int t = 0; t < S; ++t) {
    const float al = P.gates[t], et = P.gates[S + t], th = P.gates[2 * S + t];

    // ---- P1: depthwise conv (on the fly) + l2norm + bf16/transposed staging
    if (gw < 96) {
      const int role = gw >> 5;  // 0=k, 1=q, 2=v
      const int b = gw & 31;
      const float* pre = (role == 0) ? P.keys_pre : (role == 1) ? P.queries_pre : P.values_pre;
      const float* cw  = (role == 0) ? P.ck_w : (role == 1) ? P.cq_w : P.cv_w;
      const float* cb  = (role == 0) ? P.ck_b : (role == 1) ? P.cq_b : P.cv_b;
      float val[16];
#pragma unroll
      for (int ii = 0; ii < 16; ++ii) {
        const int c = ii * 32 + lane;
        float acc = cb[c];
#pragma unroll
        for (int j = 0; j < 3; ++j) {
          const int s2 = t - 1 + j;
          if (0 <= s2 && s2 < S)
            acc += cw[c * 3 + j] * pre[((size_t)s2 * B + b) * D + c];
        }
        val[ii] = acc;
      }
      if (role < 2) {
        float ss = 0.f;
#pragma unroll
        for (int ii = 0; ii < 16; ++ii) ss += val[ii] * val[ii];
        ss = wave_reduce_add(ss);
        const float scale = 1.0f / fmaxf(sqrtf(ss), 1e-12f);
#pragma unroll
        for (int ii = 0; ii < 16; ++ii) {
          const int c = ii * 32 + lane;
          const __bf16 hv = to_bf(val[ii] * scale);
          if (role == 0) { P.kt_bf[b * D + c] = hv; P.ktT_bf[c * B + b] = hv; }
          else           { P.qt_bf[b * D + c] = hv; }
        }
      } else {
#pragma unroll
        for (int ii = 0; ii < 16; ++ii) { const int c = ii * 32 + lane; P.vt[b * D + c] = val[ii]; }
      }
    } else if (gw == 96) {
      for (int e = lane; e < D; e += 32) { P.db1[e] = 0.f; P.db2[e] = 0.f; }
    }
    grid_sync(P.bar, gen);

    // ---- P2: z1 = kt @ W1^T + b1 ; h = silu(z1) ; sp = silu'(z1)
    if (gw < 64) {
      const int mt = gw >> 5, nt = gw & 31;
      v8f acc = Z8;
      for (int k0 = 0; k0 < D; k0 += 32)
        acc = wmma_bf16(load_frag_A(P.kt_bf, D, mt * 16, k0, lane),
                        load_frag_Bt(P.W1bf, D, nt * 16, k0, lane), acc);
      const int col = nt * 16 + (lane & 15);
      const int rbase = mt * 16 + ((lane >> 4) << 3);
      const float bv = P.b1[col];
#pragma unroll
      for (int r = 0; r < 8; ++r) {
        const int row = rbase + r;
        const float z = acc[r] + bv;
        const float sg = sigmoidf_(z);
        const float hv = z * sg;
        P.h_bf[row * D + col] = to_bf(hv);
        P.hT_bf[col * B + row] = to_bf(hv);
        P.sp[row * D + col] = sg * (1.0f + z * (1.0f - sg));
      }
    }
    grid_sync(P.bar, gen);

    // ---- P3: pred = h @ W2^T + b2 ; dpred = 2*(pred - v)/(B*D) ; db2 partials
    if (gw < 64) {
      const int mt = gw >> 5, nt = gw & 31;
      v8f acc = Z8;
      for (int k0 = 0; k0 < D; k0 += 32)
        acc = wmma_bf16(load_frag_A(P.h_bf, D, mt * 16, k0, lane),
                        load_frag_Bt(P.W2bf, D, nt * 16, k0, lane), acc);
      const int col = nt * 16 + (lane & 15);
      const int rbase = mt * 16 + ((lane >> 4) << 3);
      const float bv = P.b2[col];
      float dsum = 0.f;
#pragma unroll
      for (int r = 0; r < 8; ++r) {
        const int row = rbase + r;
        const float d = (acc[r] + bv - P.vt[row * D + col]) * DPRED_SCALE;
        P.dpred_bf[row * D + col] = to_bf(d);
        P.dpredT_bf[col * B + row] = to_bf(d);
        dsum += d;
      }
      atomicAdd(&P.db2[col], dsum);
    }
    grid_sync(P.bar, gen);

    // ---- P4: dh = dpred @ W2 (old, via W2bfT) -> dpre ; dW2 tiles + fused S2/W2 update ; b2 update
    if (gw < 64) {
      const int mt = gw >> 5, nt = gw & 31;
      v8f acc = Z8;
      for (int k0 = 0; k0 < D; k0 += 32)
        acc = wmma_bf16(load_frag_A(P.dpred_bf, D, mt * 16, k0, lane),
                        load_frag_Bt(P.W2bfT, D, nt * 16, k0, lane), acc);
      const int col = nt * 16 + (lane & 15);
      const int rbase = mt * 16 + ((lane >> 4) << 3);
      float dsum = 0.f;
#pragma unroll
      for (int r = 0; r < 8; ++r) {
        const int row = rbase + r;
        const float dp = acc[r] * P.sp[row * D + col];
        P.dpreT_bf[col * B + row] = to_bf(dp);
        dsum += dp;
      }
      atomicAdd(&P.db1[col], dsum);
    } else {
      for (int tile = gw - 64; tile < 1024; tile += NWAVE - 64) {
        const int jt = tile >> 5, lt = tile & 31;
        v8f acc = wmma_bf16(load_frag_A(P.dpredT_bf, B, jt * 16, 0, lane),
                            load_frag_Bt(P.hT_bf, B, lt * 16, 0, lane), Z8);
        const int col = lt * 16 + (lane & 15);
        const int rbase = jt * 16 + ((lane >> 4) << 3);
#pragma unroll
        for (int r = 0; r < 8; ++r) {
          const size_t idx = (size_t)(rbase + r) * D + col;
          const float s2n = et * P.S2[idx] - th * acc[r];
          P.S2[idx] = s2n;
          const float w2n = (1.0f - al) * P.W2[idx] + s2n;
          P.W2[idx] = w2n;
          P.W2bf[idx] = to_bf(w2n);
        }
      }
      if (gw == NWAVE - 1) {
        for (int e = lane; e < D; e += 32) {
          const float s2n = et * P.Sb2[e] - th * P.db2[e];
          P.Sb2[e] = s2n;
          P.b2[e] = (1.0f - al) * P.b2[e] + s2n;
        }
      }
    }
    grid_sync(P.bar, gen);

    // ---- P5: dW1 tiles + fused S1/W1 update ; b1 update
    for (int tile = gw; tile < 1024; tile += NWAVE) {
      const int jt = tile >> 5, lt = tile & 31;
      v8f acc = wmma_bf16(load_frag_A(P.dpreT_bf, B, jt * 16, 0, lane),
                          load_frag_Bt(P.ktT_bf, B, lt * 16, 0, lane), Z8);
      const int col = lt * 16 + (lane & 15);
      const int rbase = jt * 16 + ((lane >> 4) << 3);
#pragma unroll
      for (int r = 0; r < 8; ++r) {
        const size_t idx = (size_t)(rbase + r) * D + col;
        const float s1n = et * P.S1[idx] - th * acc[r];
        P.S1[idx] = s1n;
        const float w1n = (1.0f - al) * P.W1[idx] + s1n;
        P.W1[idx] = w1n;
        P.W1bf[idx] = to_bf(w1n);
      }
    }
    if (gw == NWAVE - 1) {
      for (int e = lane; e < D; e += 32) {
        const float s1n = et * P.Sb1[e] - th * P.db1[e];
        P.Sb1[e] = s1n;
        P.b1[e] = (1.0f - al) * P.b1[e] + s1n;
      }
    }
    grid_sync(P.bar, gen);

    // ---- P6: hq = silu(q @ W1_new^T + b1_new) ; idle waves refresh W2bfT from new W2bf
    if (gw < 64) {
      const int mt = gw >> 5, nt = gw & 31;
      v8f acc = Z8;
      for (int k0 = 0; k0 < D; k0 += 32)
        acc = wmma_bf16(load_frag_A(P.qt_bf, D, mt * 16, k0, lane),
                        load_frag_Bt(P.W1bf, D, nt * 16, k0, lane), acc);
      const int col = nt * 16 + (lane & 15);
      const int rbase = mt * 16 + ((lane >> 4) << 3);
      const float bv = P.b1[col];
#pragma unroll
      for (int r = 0; r < 8; ++r) {
        const float z = acc[r] + bv;
        P.hq_bf[(rbase + r) * D + col] = to_bf(z * sigmoidf_(z));
      }
    } else {
      for (int i = (gw - 64) * 32 + lane; i < D * D; i += (NWAVE - 64) * 32) {
        const int row = i >> 9, col = i & (D - 1);
        P.W2bfT[(size_t)col * D + row] = P.W2bf[i];
      }
    }
    grid_sync(P.bar, gen);

    // ---- P7: out[t] = hq @ W2_new^T + b2_new
    if (gw < 64) {
      const int mt = gw >> 5, nt = gw & 31;
      v8f acc = Z8;
      for (int k0 = 0; k0 < D; k0 += 32)
        acc = wmma_bf16(load_frag_A(P.hq_bf, D, mt * 16, k0, lane),
                        load_frag_Bt(P.W2bf, D, nt * 16, k0, lane), acc);
      const int col = nt * 16 + (lane & 15);
      const int rbase = mt * 16 + ((lane >> 4) << 3);
      const float bv = P.b2[col];
#pragma unroll
      for (int r = 0; r < 8; ++r)
        P.out[((size_t)t * B + rbase + r) * D + col] = acc[r] + bv;
    }
    grid_sync(P.bar, gen);
  }
}

// ---------------------------------------------------------------- launch

extern "C" void kernel_launch(void* const* d_in, const int* in_sizes, int n_in,
                              void* d_out, int out_size, void* d_ws, size_t ws_size,
                              hipStream_t stream) {
  (void)in_sizes; (void)n_in; (void)out_size; (void)ws_size;
  const float* x    = (const float*)d_in[0];
  const float* Wk_w = (const float*)d_in[1];
  const float* Wk_b = (const float*)d_in[2];
  const float* Wv_w = (const float*)d_in[3];
  const float* Wv_b = (const float*)d_in[4];
  const float* Wq_w = (const float*)d_in[5];
  const float* Wq_b = (const float*)d_in[6];
  const float* ck_w = (const float*)d_in[7];
  const float* ck_b = (const float*)d_in[8];
  const float* cv_w = (const float*)d_in[9];
  const float* cv_b = (const float*)d_in[10];
  const float* cq_w = (const float*)d_in[11];
  const float* cq_b = (const float*)d_in[12];
  const float* M_W1 = (const float*)d_in[13];
  const float* M_b1 = (const float*)d_in[14];
  const float* M_W2 = (const float*)d_in[15];
  const float* M_b2 = (const float*)d_in[16];
  const float* a_w  = (const float*)d_in[17];
  const float* a_b  = (const float*)d_in[18];
  const float* e_w  = (const float*)d_in[19];
  const float* e_b  = (const float*)d_in[20];
  const float* t_w  = (const float*)d_in[21];
  const float* t_b  = (const float*)d_in[22];

  char* p = (char*)d_ws;
  auto carve = [&](size_t bytes) -> void* {
    void* r = (void*)p;
    p += (bytes + 255) & ~(size_t)255;
    return r;
  };

  __bf16* xbf        = (__bf16*)carve((size_t)SB * D * 2);
  __bf16* Wkbf       = (__bf16*)carve((size_t)D * D * 2);
  __bf16* Wvbf       = (__bf16*)carve((size_t)D * D * 2);
  __bf16* Wqbf       = (__bf16*)carve((size_t)D * D * 2);
  float* keys_pre    = (float*)carve((size_t)SB * D * 4);
  float* values_pre  = (float*)carve((size_t)SB * D * 4);
  float* queries_pre = (float*)carve((size_t)SB * D * 4);
  float* gates       = (float*)carve((size_t)3 * S * 4);
  float* W1  = (float*)carve((size_t)D * D * 4);
  float* W2  = (float*)carve((size_t)D * D * 4);
  float* S1  = (float*)carve((size_t)D * D * 4);
  float* S2  = (float*)carve((size_t)D * D * 4);
  float* b1  = (float*)carve(D * 4);
  float* b2  = (float*)carve(D * 4);
  float* Sb1 = (float*)carve(D * 4);
  float* Sb2 = (float*)carve(D * 4);
  __bf16* W1bf  = (__bf16*)carve((size_t)D * D * 2);
  __bf16* W2bf  = (__bf16*)carve((size_t)D * D * 2);
  __bf16* W2bfT = (__bf16*)carve((size_t)D * D * 2);
  __bf16* kt_bf    = (__bf16*)carve((size_t)B * D * 2);
  __bf16* ktT_bf   = (__bf16*)carve((size_t)D * B * 2);
  __bf16* qt_bf    = (__bf16*)carve((size_t)B * D * 2);
  __bf16* h_bf     = (__bf16*)carve((size_t)B * D * 2);
  __bf16* hT_bf    = (__bf16*)carve((size_t)D * B * 2);
  __bf16* dpred_bf  = (__bf16*)carve((size_t)B * D * 2);
  __bf16* dpredT_bf = (__bf16*)carve((size_t)D * B * 2);
  __bf16* dpreT_bf  = (__bf16*)carve((size_t)D * B * 2);
  __bf16* hq_bf     = (__bf16*)carve((size_t)B * D * 2);
  float* vt  = (float*)carve((size_t)B * D * 4);
  float* sp  = (float*)carve((size_t)B * D * 4);
  float* db1 = (float*)carve(D * 4);
  float* db2 = (float*)carve(D * 4);
  unsigned* bar = (unsigned*)carve(256);

  (void)hipMemsetAsync(gates, 0, (size_t)3 * S * sizeof(float), stream);
  (void)hipMemsetAsync(bar, 0, 256, stream);

  f2bf_kernel<<<4096, 256, 0, stream>>>(x, xbf, (size_t)SB * D);
  f2bf_kernel<<<256, 256, 0, stream>>>(Wk_w, Wkbf, (size_t)D * D);
  f2bf_kernel<<<256, 256, 0, stream>>>(Wv_w, Wvbf, (size_t)D * D);
  f2bf_kernel<<<256, 256, 0, stream>>>(Wq_w, Wqbf, (size_t)D * D);

  const int gblk = (SB / 128) * (D / 16);  // 8192 blocks: 128x16 out per block
  gemm_bias_wt<<<gblk, 256, 0, stream>>>(xbf, Wkbf, Wk_b, keys_pre, SB, D, D);
  gemm_bias_wt<<<gblk, 256, 0, stream>>>(xbf, Wvbf, Wv_b, values_pre, SB, D, D);
  gemm_bias_wt<<<gblk, 256, 0, stream>>>(xbf, Wqbf, Wq_b, queries_pre, SB, D, D);

  gate_kernel<<<SB / WPB, 256, 0, stream>>>(x, a_w, a_b, e_w, e_b, t_w, t_b, gates);

  init_state<<<1024, 256, 0, stream>>>(M_W1, M_b1, M_W2, M_b2,
                                       W1, b1, W2, b2, S1, Sb1, S2, Sb2,
                                       W1bf, W2bf, W2bfT);

  ScanArgs SA;
  SA.keys_pre = keys_pre; SA.values_pre = values_pre; SA.queries_pre = queries_pre;
  SA.ck_w = ck_w; SA.ck_b = ck_b; SA.cv_w = cv_w; SA.cv_b = cv_b; SA.cq_w = cq_w; SA.cq_b = cq_b;
  SA.gates = gates;
  SA.W1 = W1; SA.W2 = W2; SA.S1 = S1; SA.S2 = S2;
  SA.b1 = b1; SA.b2 = b2; SA.Sb1 = Sb1; SA.Sb2 = Sb2;
  SA.W1bf = W1bf; SA.W2bf = W2bf; SA.W2bfT = W2bfT;
  SA.kt_bf = kt_bf; SA.ktT_bf = ktT_bf; SA.qt_bf = qt_bf;
  SA.h_bf = h_bf; SA.hT_bf = hT_bf;
  SA.dpred_bf = dpred_bf; SA.dpredT_bf = dpredT_bf; SA.dpreT_bf = dpreT_bf;
  SA.hq_bf = hq_bf;
  SA.vt = vt; SA.sp = sp; SA.db1 = db1; SA.db2 = db2;
  SA.bar = bar;
  SA.out = (float*)d_out;

  scan_kernel<<<NBLK, 256, 0, stream>>>(SA);
}
